// PointCloudCompletionNetwork_84413287236169
// MI455X (gfx1250) — compile-verified, matrix operand order checked by reading, not run on hardware
//
#include <hip/hip_runtime.h>
#include <hip/hip_bf16.h>

// ---------------------------------------------------------------------------
// PointCloudCompletionNetwork on gfx1250 (CDNA5, wave32, WMMA 16x16x32 f16)
//
// Pipeline:
//   1) voxelize points -> (64,1,32,32,32) occupancy grid
//   2) 3x conv3d stride2 + relu (direct, tiny FLOPs)
//   3) FC chain via f16 WMMA TN-GEMM (C = A * B^T, K-contiguous operands)
//   4) per patch p in 0..3:
//        linear 128->4096 (WMMA GEMM)
//        4x convT2d(k4,s2,p1) as [repack W -> f16 A(M=OC*16,K=IC)]
//                               [transpose-cast X -> f16 Xt(N=B*S,K=IC)]
//                               [WMMA GEMM -> col(M,N) f32]
//                               [col2im gather + bias + relu]
//        avgpool2x2 + 1x1 conv head -> out[b, p*1024 + hw, 0..2]
// ---------------------------------------------------------------------------

typedef __attribute__((ext_vector_type(16))) _Float16 v16h;
typedef __attribute__((ext_vector_type(8)))  _Float16 v8h;
typedef __attribute__((ext_vector_type(8)))  float    v8f;

#define CDIV(a, b) (((a) + (b) - 1) / (b))

// ---------------------------------------------------------------- utilities
__global__ void pcn_fill_zero(float* __restrict__ p, int n) {
    int i = blockIdx.x * blockDim.x + threadIdx.x;
    if (i < n) p[i] = 0.0f;
}

__global__ void pcn_voxelize(const float* __restrict__ pc, float* __restrict__ grid,
                             int B, int N) {
    int i = blockIdx.x * blockDim.x + threadIdx.x;
    if (i >= B * N) return;
    int b = i / N;
    const float* p = pc + (size_t)i * 3;
    int x = (int)((p[0] + 1.0f) * 0.5f * 31.0f);
    int y = (int)((p[1] + 1.0f) * 0.5f * 31.0f);
    int z = (int)((p[2] + 1.0f) * 0.5f * 31.0f);
    x = min(max(x, 0), 31); y = min(max(y, 0), 31); z = min(max(z, 0), 31);
    grid[(((size_t)b * 32 + z) * 32 + y) * 32 + x] = 1.0f;   // idempotent store
}

// direct conv3d, stride 2, pad 1, k=3, + relu.  X:(B,Cin,Din^3) W:(Cout,Cin,27)
__global__ void pcn_conv3d_s2_relu(const float* __restrict__ X, const float* __restrict__ W,
                                   const float* __restrict__ bias, float* __restrict__ Y,
                                   int B, int Cin, int Din, int Cout, int Dout) {
    int idx = blockIdx.x * blockDim.x + threadIdx.x;
    int total = B * Cout * Dout * Dout * Dout;
    if (idx >= total) return;
    int ox = idx % Dout; int t = idx / Dout;
    int oy = t % Dout;   t /= Dout;
    int oz = t % Dout;   t /= Dout;
    int oc = t % Cout;   int b = t / Cout;
    float acc = bias[oc];
    for (int ic = 0; ic < Cin; ++ic) {
        const float* xp = X + ((size_t)(b * Cin + ic)) * Din * Din * Din;
        const float* wp = W + ((size_t)(oc * Cin + ic)) * 27;
        for (int kz = 0; kz < 3; ++kz) {
            int iz = oz * 2 - 1 + kz; if ((unsigned)iz >= (unsigned)Din) continue;
            for (int ky = 0; ky < 3; ++ky) {
                int iy = oy * 2 - 1 + ky; if ((unsigned)iy >= (unsigned)Din) continue;
                for (int kx = 0; kx < 3; ++kx) {
                    int ix = ox * 2 - 1 + kx; if ((unsigned)ix >= (unsigned)Din) continue;
                    acc += xp[((size_t)iz * Din + iy) * Din + ix] * wp[(kz * 3 + ky) * 3 + kx];
                }
            }
        }
    }
    Y[idx] = fmaxf(acc, 0.0f);
}

// Y[r*cols + c] = (f16) X[r*ldx + c]   (plain cast when ldx == cols)
__global__ void pcn_slice_cast(const float* __restrict__ X, _Float16* __restrict__ Y,
                               int rows, int cols, int ldx) {
    int i = blockIdx.x * blockDim.x + threadIdx.x;
    if (i >= rows * cols) return;
    int c = i % cols, r = i / cols;
    Y[i] = (_Float16)X[(size_t)r * ldx + c];
}

// X:(B,C,S) f32 -> Xt:(B*S, C) f16
__global__ void pcn_transpose_cast(const float* __restrict__ X, _Float16* __restrict__ Xt,
                                   int B, int C, int S) {
    int i = blockIdx.x * blockDim.x + threadIdx.x;
    if (i >= B * C * S) return;
    int s = i % S; int t = i / S; int c = t % C; int b = t / C;
    Xt[((size_t)b * S + s) * C + c] = (_Float16)X[i];
}

// W:(IC,OC,4,4) f32 -> A:(OC*16, IC) f16,  A[(oc*16+t)*IC + ic] = W[ic][oc][t]
__global__ void pcn_repack_wt(const float* __restrict__ W, _Float16* __restrict__ A,
                              int IC, int OC) {
    int i = blockIdx.x * blockDim.x + threadIdx.x;
    if (i >= IC * OC * 16) return;
    int t = i % 16; int r = i / 16; int oc = r % OC; int ic = r / OC;
    A[((size_t)(oc * 16 + t)) * IC + ic] = (_Float16)W[i];
}

// ------------------------------------------------------------- WMMA TN GEMM
// C[M,N] (f32) = A[M,K] (f16, K-contig) * B[N,K]^T (f16, K-contig)
// optional bias along N, optional relu.  Requires M%16==0, N%64==0, K%32==0.
// One wave -> 16x64 strip of C (A fragment reused across 4 WMMAs per K step).
#define WMMA16x16x32(a, b, c) \
    __builtin_amdgcn_wmma_f32_16x16x32_f16(false, (a), false, (b), (short)0, (c), false, false)

__global__ __launch_bounds__(256)
void pcn_wmma_gemm_tn(const _Float16* __restrict__ A, const _Float16* __restrict__ Bm,
                      float* __restrict__ C, const float* __restrict__ bias,
                      int M, int N, int K, int relu) {
    const int lane    = threadIdx.x & 31;
    const int wid     = blockIdx.x * (blockDim.x >> 5) + (threadIdx.x >> 5);
    const int tilesN4 = N >> 6;
    const int tiles   = (M >> 4) * tilesN4;
    if (wid >= tiles) return;
    const int tm = (wid / tilesN4) << 4;
    const int tn = (wid % tilesN4) << 6;

    const int r  = lane & 15;   // matrix row within tile (A: M-row, B: N-row)
    const int hi = lane >> 4;   // K-half selector for A/B fragments

    const _Float16* arow = A  + (size_t)(tm + r) * K;
    const _Float16* b0   = Bm + (size_t)(tn + r) * K;

    v8f acc0 = {}, acc1 = {}, acc2 = {}, acc3 = {};

    for (int k0 = 0; k0 < K; k0 += 32) {
        const int o1 = k0 + (hi << 3);   // halfs [o1, o1+8)   -> frag elems 0..7
        const int o2 = o1 + 16;          // halfs [o2, o2+8)   -> frag elems 8..15
        v16h a;
        *((v8h*)&a)     = *(const v8h*)(arow + o1);
        *((v8h*)&a + 1) = *(const v8h*)(arow + o2);

        v16h b;
        const _Float16* br;
        br = b0;                     // N-tile 0
        *((v8h*)&b)     = *(const v8h*)(br + o1);
        *((v8h*)&b + 1) = *(const v8h*)(br + o2);
        acc0 = WMMA16x16x32(a, b, acc0);

        br = b0 + (size_t)16 * K;    // N-tile 1
        *((v8h*)&b)     = *(const v8h*)(br + o1);
        *((v8h*)&b + 1) = *(const v8h*)(br + o2);
        acc1 = WMMA16x16x32(a, b, acc1);

        br = b0 + (size_t)32 * K;    // N-tile 2
        *((v8h*)&b)     = *(const v8h*)(br + o1);
        *((v8h*)&b + 1) = *(const v8h*)(br + o2);
        acc2 = WMMA16x16x32(a, b, acc2);

        br = b0 + (size_t)48 * K;    // N-tile 3
        *((v8h*)&b)     = *(const v8h*)(br + o1);
        *((v8h*)&b + 1) = *(const v8h*)(br + o2);
        acc3 = WMMA16x16x32(a, b, acc3);
    }

    // C/D layout: elem e of lane L -> m = tm + e + 8*(L/16), n = tile_n + (L&15)
    const int mBase = tm + (hi << 3);
    #pragma unroll
    for (int t = 0; t < 4; ++t) {
        const v8f& acc = (t == 0) ? acc0 : (t == 1) ? acc1 : (t == 2) ? acc2 : acc3;
        const int n = tn + t * 16 + r;
        const float bv = bias ? bias[n] : 0.0f;
        #pragma unroll
        for (int e = 0; e < 8; ++e) {
            float v = acc[e] + bv;
            if (relu) v = fmaxf(v, 0.0f);
            C[(size_t)(mBase + e) * N + n] = v;
        }
    }
}

// --------------------------------------------------------------- col2im
// col:(OC*16, B*IH*IW) f32 -> Y:(B,OC,2IH,2IW) f32, gather <=4 taps, +bias, relu
__global__ void pcn_col2im_bias_relu(const float* __restrict__ col, const float* __restrict__ bias,
                                     float* __restrict__ Y, int B, int OC, int IH, int IW) {
    const int OH = 2 * IH, OW = 2 * IW;
    int idx = blockIdx.x * blockDim.x + threadIdx.x;
    int total = B * OC * OH * OW;
    if (idx >= total) return;
    int ox = idx % OW; int t = idx / OW;
    int oy = t % OH;   t /= OH;
    int oc = t % OC;   int b = t / OC;
    const int N = B * IH * IW;
    float acc = bias[oc];
    for (int ky = (oy + 1) & 1; ky < 4; ky += 2) {
        int ih = (oy + 1 - ky) >> 1;
        if ((unsigned)ih >= (unsigned)IH) continue;
        for (int kx = (ox + 1) & 1; kx < 4; kx += 2) {
            int iw = (ox + 1 - kx) >> 1;
            if ((unsigned)iw >= (unsigned)IW) continue;
            int m = oc * 16 + ky * 4 + kx;
            int n = (b * IH + ih) * IW + iw;
            acc += col[(size_t)m * N + n];
        }
    }
    Y[idx] = fmaxf(acc, 0.0f);
}

// ------------------------------------------------------------ pool + head
// Y:(B,32,64,64), avgpool 2x2 -> (B,32,32,32), 1x1 conv 32->3,
// out[b, p*1024 + h*32 + w, o]  (out shape (B,4096,3))
__global__ void pcn_pool_head(const float* __restrict__ Y, const float* __restrict__ rxw,
                              const float* __restrict__ rxb, float* __restrict__ out,
                              int B, int p) {
    int idx = blockIdx.x * blockDim.x + threadIdx.x;
    if (idx >= B * 32 * 32) return;
    int w = idx % 32; int h = (idx / 32) % 32; int b = idx / 1024;
    float pooled[32];
    #pragma unroll
    for (int c = 0; c < 32; ++c) {
        const float* yp = Y + (((size_t)(b * 32 + c) * 64 + 2 * h) * 64 + 2 * w);
        pooled[c] = 0.25f * (yp[0] + yp[1] + yp[64] + yp[65]);
    }
    for (int o = 0; o < 3; ++o) {
        float acc = rxb[p * 3 + o];
        const float* wrow = rxw + (size_t)(p * 3 + o) * 32;
        #pragma unroll
        for (int c = 0; c < 32; ++c) acc += pooled[c] * wrow[c];
        out[((size_t)b * 4096 + p * 1024 + h * 32 + w) * 3 + o] = acc;
    }
}

// ===========================================================================
extern "C" void kernel_launch(void* const* d_in, const int* in_sizes, int n_in,
                              void* d_out, int out_size, void* d_ws, size_t ws_size,
                              hipStream_t stream) {
    (void)in_sizes; (void)n_in; (void)out_size; (void)ws_size;
    const int B = 64, NPTS = 32768;
    const int TB = 256;

    const float* partial = (const float*)d_in[0];
    const float* ew1 = (const float*)d_in[1];  const float* eb1 = (const float*)d_in[2];
    const float* ew2 = (const float*)d_in[3];  const float* eb2 = (const float*)d_in[4];
    const float* ew3 = (const float*)d_in[5];  const float* eb3 = (const float*)d_in[6];
    const float* efw = (const float*)d_in[7];  const float* efb = (const float*)d_in[8];
    const float* bw  = (const float*)d_in[9];  const float* bb  = (const float*)d_in[10];
    const float* pw  = (const float*)d_in[11]; const float* pb  = (const float*)d_in[12];
    const float* riw = (const float*)d_in[13]; const float* rib = (const float*)d_in[14];
    const float* r1w = (const float*)d_in[15]; const float* r1b = (const float*)d_in[16];
    const float* r2w = (const float*)d_in[17]; const float* r2b = (const float*)d_in[18];
    const float* r3w = (const float*)d_in[19]; const float* r3b = (const float*)d_in[20];
    const float* r4w = (const float*)d_in[21]; const float* r4b = (const float*)d_in[22];
    const float* rxw = (const float*)d_in[23]; const float* rxb = (const float*)d_in[24];
    float* out = (float*)d_out;

    // ---- workspace bump allocator (deterministic layout each call) ----
    char* ws = (char*)d_ws;
    size_t off = 0;
    auto alloc = [&](size_t bytes) -> void* {
        off = (off + 255) & ~(size_t)255;
        void* p = ws + off;
        off += bytes;
        return p;
    };

    float*    grid    = (float*)   alloc((size_t)B * 32768 * 4);          // (B,32,32,32)
    float*    f1      = (float*)   alloc((size_t)B * 8  * 4096 * 4);      // (B,8,16^3)
    float*    f2      = (float*)   alloc((size_t)B * 16 * 512  * 4);      // (B,16,8^3)
    float*    f3      = (float*)   alloc((size_t)B * 32 * 64   * 4);      // (B,32,4^3)
    _Float16* f3h     = (_Float16*)alloc((size_t)B * 2048 * 2);
    _Float16* efw16   = (_Float16*)alloc((size_t)128 * 2048 * 2);
    float*    latent  = (float*)   alloc((size_t)B * 128 * 4);
    _Float16* lat16   = (_Float16*)alloc((size_t)B * 128 * 2);
    _Float16* bw16    = (_Float16*)alloc((size_t)256 * 128 * 2);
    float*    lat2    = (float*)   alloc((size_t)B * 256 * 4);
    _Float16* lat2_16 = (_Float16*)alloc((size_t)B * 256 * 2);
    _Float16* pw16    = (_Float16*)alloc((size_t)512 * 256 * 2);
    float*    plv     = (float*)   alloc((size_t)B * 512 * 4);
    _Float16* latp16  = (_Float16*)alloc((size_t)B * 128 * 2);
    _Float16* riw16   = (_Float16*)alloc((size_t)4096 * 128 * 2);
    float*    ft0     = (float*)   alloc((size_t)B * 4096 * 4);           // (B,256,4,4)
    _Float16* wa16    = (_Float16*)alloc((size_t)2048 * 256 * 2);         // max repacked W
    _Float16* xt16    = (_Float16*)alloc((size_t)B * 1024 * 64 * 2);      // max (65536,64)
    float*    colbuf  = (float*)   alloc((size_t)512 * 65536 * 4);        // max 128 MB
    float*    bufA    = (float*)   alloc((size_t)B * 64 * 1024 * 4);      // Y1 / Y3
    float*    bufB    = (float*)   alloc((size_t)B * 32 * 4096 * 4);      // Y2 / Y4

    auto gemm = [&](const _Float16* A, const _Float16* Bm, float* C, const float* bias,
                    int M, int N, int K, int relu) {
        int waves = (M / 16) * (N / 64);
        pcn_wmma_gemm_tn<<<CDIV(waves, 8), TB, 0, stream>>>(A, Bm, C, bias, M, N, K, relu);
    };

    // ---- 1) voxelize ----
    {
        int n = B * 32768;
        pcn_fill_zero<<<CDIV(n, TB), TB, 0, stream>>>(grid, n);
        pcn_voxelize<<<CDIV(B * NPTS, TB), TB, 0, stream>>>(partial, grid, B, NPTS);
    }

    // ---- 2) encoder convs ----
    pcn_conv3d_s2_relu<<<CDIV(B * 8 * 4096, TB), TB, 0, stream>>>(grid, ew1, eb1, f1, B, 1,  32, 8,  16);
    pcn_conv3d_s2_relu<<<CDIV(B * 16 * 512, TB), TB, 0, stream>>>(f1,   ew2, eb2, f2, B, 8,  16, 16, 8);
    pcn_conv3d_s2_relu<<<CDIV(B * 32 * 64,  TB), TB, 0, stream>>>(f2,   ew3, eb3, f3, B, 16, 8,  32, 4);

    // ---- 3) FC chain ----
    pcn_slice_cast<<<CDIV(B * 2048, TB), TB, 0, stream>>>(f3, f3h, B, 2048, 2048);
    pcn_slice_cast<<<CDIV(128 * 2048, TB), TB, 0, stream>>>(efw, efw16, 128, 2048, 2048);
    gemm(f3h, efw16, latent, efb, B, 128, 2048, 0);

    pcn_slice_cast<<<CDIV(B * 128, TB), TB, 0, stream>>>(latent, lat16, B, 128, 128);
    pcn_slice_cast<<<CDIV(256 * 128, TB), TB, 0, stream>>>(bw, bw16, 256, 128, 128);
    gemm(lat16, bw16, lat2, bb, B, 256, 128, 1);

    pcn_slice_cast<<<CDIV(B * 256, TB), TB, 0, stream>>>(lat2, lat2_16, B, 256, 256);
    pcn_slice_cast<<<CDIV(512 * 256, TB), TB, 0, stream>>>(pw, pw16, 512, 256, 256);
    gemm(lat2_16, pw16, plv, pb, B, 512, 256, 1);

    // ---- 4) per-patch decoder ----
    for (int p = 0; p < 4; ++p) {
        // patch latent slice (B,128) from plv (B,512), then 128->4096 linear
        pcn_slice_cast<<<CDIV(B * 128, TB), TB, 0, stream>>>(plv + p * 128, latp16, B, 128, 512);
        pcn_slice_cast<<<CDIV(4096 * 128, TB), TB, 0, stream>>>(riw + (size_t)p * 4096 * 128,
                                                                riw16, 4096, 128, 128);
        gemm(latp16, riw16, ft0, rib + p * 4096, B, 4096, 128, 0);  // (B,256,4,4)

        struct { int IC, OC, IH; const float* W; const float* bias; const float* X; float* Y; }
        L[4] = {
            {256, 128, 4,  r1w + (size_t)p * 256 * 128 * 16, r1b + p * 128, ft0,  bufA},
            {128, 64,  8,  r2w + (size_t)p * 128 * 64  * 16, r2b + p * 64,  bufA, bufB},
            {64,  64,  16, r3w + (size_t)p * 64  * 64  * 16, r3b + p * 64,  bufB, bufA},
            {64,  32,  32, r4w + (size_t)p * 64  * 32  * 16, r4b + p * 32,  bufA, bufB},
        };
        for (int l = 0; l < 4; ++l) {
            const int IC = L[l].IC, OC = L[l].OC, IH = L[l].IH, S = IH * IH;
            const int M = OC * 16, N = B * S, K = IC;
            pcn_repack_wt<<<CDIV(IC * OC * 16, TB), TB, 0, stream>>>(L[l].W, wa16, IC, OC);
            pcn_transpose_cast<<<CDIV(B * IC * S, TB), TB, 0, stream>>>(L[l].X, xt16, B, IC, S);
            gemm(wa16, xt16, colbuf, nullptr, M, N, K, 0);
            pcn_col2im_bias_relu<<<CDIV(B * OC * 4 * S, TB), TB, 0, stream>>>(
                colbuf, L[l].bias, L[l].Y, B, OC, IH, IH);
        }

        pcn_pool_head<<<CDIV(B * 1024, TB), TB, 0, stream>>>(bufB, rxw, rxb, out, B, p);
    }
}
